// SpatialAttBlock_6330781794411
// MI455X (gfx1250) — matrix operand order repaired and checked by reading, not compiled, not run on hardware
//
#include <hip/hip_runtime.h>
#include <hip/hip_bf16.h>

// MI455X / gfx1250, wave32. Attention rewritten via associativity:
//   out = x_opt + x_sar + g_o * Wp_o @ (A_opt @ (A_sar @ v_opt))
//                       + g_s * Wp_s @ (A_opt @ (A_sar @ v_sar))
// => two chained flash-attention passes, all matmuls on v_wmma_f32_16x16x32_f16.
// 64-key chunks: 4 S^T WMMAs + 8 PV WMMAs per iteration, B-operand reuse hints
// on same-B runs.

typedef _Float16 half8_t  __attribute__((ext_vector_type(8)));
typedef _Float16 half16_t __attribute__((ext_vector_type(16)));
typedef float    float8_t __attribute__((ext_vector_type(8)));

union Frag16 { half16_t h16; half8_t h8[2]; unsigned int u[8]; };
union F8     { float8_t v; float f[8]; };
union H2     { unsigned int u; _Float16 h[2]; };

#define LSEQ 4096   // H*W
#define NB   2      // batch

// ---------------------------------------------------------------------------
// Stage A: QKV 1x1 convs. Q,K row-major [N][L][32] f16; V transposed [N][64][L]
// (rows 0..31 = v_opt channels, 32..63 = v_sar channels).
// ---------------------------------------------------------------------------
__global__ void qkv_kernel(const float* __restrict__ x_opt,
                           const float* __restrict__ x_sar,
                           const float* __restrict__ wq_opt, const float* __restrict__ wk_opt,
                           const float* __restrict__ wv_opt,
                           const float* __restrict__ wq_sar, const float* __restrict__ wk_sar,
                           const float* __restrict__ wv_sar,
                           _Float16* __restrict__ Q1, _Float16* __restrict__ K1,
                           _Float16* __restrict__ Vt1,
                           _Float16* __restrict__ Q2, _Float16* __restrict__ K2)
{
    __shared__ float xt[64][128];
    const int n   = blockIdx.y;
    const int tid = threadIdx.x;
    const int l   = blockIdx.x * 128 + tid;

    const float* xs[2] = { x_opt, x_sar };
    const float* wq[2] = { wq_opt, wq_sar };
    const float* wk[2] = { wk_opt, wk_sar };
    const float* wv[2] = { wv_opt, wv_sar };
    _Float16* qo[2] = { Q2, Q1 };   // mod0 = opt -> pass 2
    _Float16* ko[2] = { K2, K1 };

    for (int mod = 0; mod < 2; ++mod) {
        const float* xb = xs[mod] + (size_t)n * 64 * LSEQ + l;
        __syncthreads();
        for (int c = 0; c < 64; ++c) xt[c][tid] = xb[(size_t)c * LSEQ];
        __syncthreads();
        for (int o = 0; o < 32; ++o) {
            float aq = 0.f, ak = 0.f, av = 0.f;
            const float* wqr = wq[mod] + o * 64;
            const float* wkr = wk[mod] + o * 64;
            const float* wvr = wv[mod] + o * 64;
            #pragma unroll 8
            for (int c = 0; c < 64; ++c) {
                const float xv = xt[c][tid];
                aq += wqr[c] * xv;
                ak += wkr[c] * xv;
                av += wvr[c] * xv;
            }
            qo[mod][(size_t)n * LSEQ * 32 + (size_t)l * 32 + o] = (_Float16)aq;
            ko[mod][(size_t)n * LSEQ * 32 + (size_t)l * 32 + o] = (_Float16)ak;
            Vt1[(size_t)n * 64 * LSEQ + (size_t)(mod * 32 + o) * LSEQ + l] = (_Float16)av;
        }
    }
}

// ---------------------------------------------------------------------------
// Stage B: transposed flash attention, one wave per 16-query tile, 64 keys
// per iteration.
//   S^T(16k x 16q) = K_tile(16x32) x Q^T(32x16)      -> 4 WMMA / 64 keys
//   O^T(64ch x16q) += V^T_tile(16x32) x P^T(32x16)   -> 8 WMMA / 64 keys
// Softmax runs down M (keys): per-query stats are per-lane scalars
// (one shfl_xor(16) merges the two half-wave row groups).
// Output O^T is written in the V^T layout [64][L] -> pass2 consumes pass1.
// ---------------------------------------------------------------------------
__global__ void attn_kernel(const _Float16* __restrict__ Q,
                            const _Float16* __restrict__ K,
                            const _Float16* __restrict__ Vt,
                            _Float16* __restrict__ Ot)
{
    const int lane = threadIdx.x & 31;
    const int wave = threadIdx.x >> 5;
    const int g    = lane >> 4;      // half-wave group
    const int ln   = lane & 15;
    const int n    = blockIdx.y;
    const int m0   = (blockIdx.x * 4 + wave) * 16;   // query tile base

    const _Float16* Qn = Q  + (size_t)n * LSEQ * 32;
    const _Float16* Kn = K  + (size_t)n * LSEQ * 32;
    const _Float16* Vn = Vt + (size_t)n * 64 * LSEQ;
    _Float16*       On = Ot + (size_t)n * 64 * LSEQ;

    // B operand: Q^T (32x16). Lane (g,ln): h[0..7] = Q[m0+ln][8g + j], h[8..15] = Q[m0+ln][16 + 8g + j].
    Frag16 bq;
    {
        const _Float16* qr = Qn + (size_t)(m0 + ln) * 32 + g * 8;
        bq.h8[0] = *(const half8_t*)(qr);
        bq.h8[1] = *(const half8_t*)(qr + 16);
    }

    F8 oacc[4];
    #pragma unroll
    for (int t = 0; t < 4; ++t)
        #pragma unroll
        for (int r = 0; r < 8; ++r) oacc[t].f[r] = 0.f;

    float mrun = -1e30f, lrun = 0.f;
    const float8_t zero8 = {0.f,0.f,0.f,0.f,0.f,0.f,0.f,0.f};

    for (int kb = 0; kb < LSEQ; kb += 64) {
        // A operands: K rows (A 16x32 layout: g=0 -> ch 0..7/16..23, g=1 -> 8..15/24..31)
        Frag16 ak[4];
        #pragma unroll
        for (int s = 0; s < 4; ++s) {
            const _Float16* kr = Kn + (size_t)(kb + 16 * s + ln) * 32;
            ak[s].h8[0] = *(const half8_t*)(kr + 8 * g);
            ak[s].h8[1] = *(const half8_t*)(kr + 16 + 8 * g);
        }
        __builtin_prefetch((const void*)(Kn + (size_t)(kb + 64 + ln) * 32), 0, 3);

        // S^T tiles, C layout: key = kb + 16s + r + 8g, query col = ln.
        // Same B operand (bq) for all 4 -> reuse_b on trailing 3.
        F8 sc[4];
        sc[0].v = __builtin_amdgcn_wmma_f32_16x16x32_f16(false, ak[0].h16, false, bq.h16,
                                                         (short)0, zero8, false, false);
        sc[1].v = __builtin_amdgcn_wmma_f32_16x16x32_f16(false, ak[1].h16, false, bq.h16,
                                                         (short)0, zero8, false, true);
        sc[2].v = __builtin_amdgcn_wmma_f32_16x16x32_f16(false, ak[2].h16, false, bq.h16,
                                                         (short)0, zero8, false, true);
        sc[3].v = __builtin_amdgcn_wmma_f32_16x16x32_f16(false, ak[3].h16, false, bq.h16,
                                                         (short)0, zero8, false, true);

        // ---- online softmax over the key (M) dimension, per query column ----
        float mnew = mrun;
        #pragma unroll
        for (int s = 0; s < 4; ++s)
            #pragma unroll
            for (int r = 0; r < 8; ++r) mnew = fmaxf(mnew, sc[s].f[r]);
        mnew = fmaxf(mnew, __shfl_xor(mnew, 16, 32));

        const float scale = __expf(mrun - mnew);
        lrun *= scale;
        #pragma unroll
        for (int t = 0; t < 4; ++t)
            #pragma unroll
            for (int r = 0; r < 8; ++r) oacc[t].f[r] *= scale;

        float p[4][8], psum = 0.f;
        #pragma unroll
        for (int s = 0; s < 4; ++s)
            #pragma unroll
            for (int r = 0; r < 8; ++r) {
                p[s][r] = __expf(sc[s].f[r] - mnew);
                psum += p[s][r];
            }
        psum += __shfl_xor(psum, 16, 32);
        lrun += psum;
        mrun  = mnew;

        // ---- build P^T B-fragments (32 keys x 16 queries each) via half-swap ----
        // bp0: keys kb+0..31 (tiles 0,1);  bp1: keys kb+32..63 (tiles 2,3).
        // g=0 lane: keys 0..7 own / 8..15 partner; g=1: 16..23 partner / 24..31 own.
        Frag16 bp0, bp1;
        #pragma unroll
        for (int k = 0; k < 4; ++k) {
            H2 a, b, c, d;
            a.h[0] = (_Float16)p[0][2*k]; a.h[1] = (_Float16)p[0][2*k + 1];
            b.h[0] = (_Float16)p[1][2*k]; b.h[1] = (_Float16)p[1][2*k + 1];
            c.h[0] = (_Float16)p[2][2*k]; c.h[1] = (_Float16)p[2][2*k + 1];
            d.h[0] = (_Float16)p[3][2*k]; d.h[1] = (_Float16)p[3][2*k + 1];
            const unsigned int oa = __shfl_xor(a.u, 16, 32);
            const unsigned int ob = __shfl_xor(b.u, 16, 32);
            const unsigned int oc = __shfl_xor(c.u, 16, 32);
            const unsigned int od = __shfl_xor(d.u, 16, 32);
            bp0.u[k]     = g ? ob  : a.u;
            bp0.u[4 + k] = g ? b.u : oa;
            bp1.u[k]     = g ? od  : c.u;
            bp1.u[4 + k] = g ? d.u : oc;
        }

        // ---- O^T += V^T x P^T: 4 channel tiles x 2 key halves ----
        Frag16 av0[4], av1[4];
        #pragma unroll
        for (int t = 0; t < 4; ++t) {
            const _Float16* vr = Vn + (size_t)(16 * t + ln) * LSEQ + kb;
            av0[t].h8[0] = *(const half8_t*)(vr + 16 * g);
            av0[t].h8[1] = *(const half8_t*)(vr + 8 + 16 * g);
            av1[t].h8[0] = *(const half8_t*)(vr + 32 + 16 * g);
            av1[t].h8[1] = *(const half8_t*)(vr + 40 + 16 * g);
        }
        // Same B (bp0) for 4 in a row -> reuse_b on trailing 3; then bp1 run.
        oacc[0].v = __builtin_amdgcn_wmma_f32_16x16x32_f16(false, av0[0].h16, false, bp0.h16,
                                                           (short)0, oacc[0].v, false, false);
        oacc[1].v = __builtin_amdgcn_wmma_f32_16x16x32_f16(false, av0[1].h16, false, bp0.h16,
                                                           (short)0, oacc[1].v, false, true);
        oacc[2].v = __builtin_amdgcn_wmma_f32_16x16x32_f16(false, av0[2].h16, false, bp0.h16,
                                                           (short)0, oacc[2].v, false, true);
        oacc[3].v = __builtin_amdgcn_wmma_f32_16x16x32_f16(false, av0[3].h16, false, bp0.h16,
                                                           (short)0, oacc[3].v, false, true);
        oacc[0].v = __builtin_amdgcn_wmma_f32_16x16x32_f16(false, av1[0].h16, false, bp1.h16,
                                                           (short)0, oacc[0].v, false, false);
        oacc[1].v = __builtin_amdgcn_wmma_f32_16x16x32_f16(false, av1[1].h16, false, bp1.h16,
                                                           (short)0, oacc[1].v, false, true);
        oacc[2].v = __builtin_amdgcn_wmma_f32_16x16x32_f16(false, av1[2].h16, false, bp1.h16,
                                                           (short)0, oacc[2].v, false, true);
        oacc[3].v = __builtin_amdgcn_wmma_f32_16x16x32_f16(false, av1[3].h16, false, bp1.h16,
                                                           (short)0, oacc[3].v, false, true);
    }

    const float inv = 1.f / lrun;
    #pragma unroll
    for (int t = 0; t < 4; ++t)
        #pragma unroll
        for (int r = 0; r < 8; ++r) {
            const int ch = 16 * t + r + 8 * g;                 // C layout row
            On[(size_t)ch * LSEQ + m0 + ln] = (_Float16)(oacc[t].f[r] * inv);
        }
}

// ---------------------------------------------------------------------------
// Stage C: out = x_opt + x_sar + g_o * Wp_o @ s_opt + g_s * Wp_s @ s_sar
// St layout [N][64][L]: rows 0..31 = s_opt^T, rows 32..63 = s_sar^T.
// ---------------------------------------------------------------------------
__global__ void proj_kernel(const float* __restrict__ x_opt, const float* __restrict__ x_sar,
                            const float* __restrict__ wp_opt, const float* __restrict__ wp_sar,
                            const float* __restrict__ g_opt, const float* __restrict__ g_sar,
                            const _Float16* __restrict__ St, float* __restrict__ out)
{
    const int n = blockIdx.z, o = blockIdx.y;
    const int l = blockIdx.x * 256 + threadIdx.x;
    const float go = g_opt[0], gs = g_sar[0];
    const _Float16* S0 = St + (size_t)n * 64 * LSEQ + l;
    float ao = 0.f, as = 0.f;
    #pragma unroll 8
    for (int c = 0; c < 32; ++c) {
        ao += wp_opt[o * 32 + c] * (float)S0[(size_t)c * LSEQ];
        as += wp_sar[o * 32 + c] * (float)S0[(size_t)(32 + c) * LSEQ];
    }
    const size_t idx = (size_t)n * 64 * LSEQ + (size_t)o * LSEQ + l;
    out[idx] = x_opt[idx] + x_sar[idx] + go * ao + gs * as;
}

// ---------------------------------------------------------------------------
extern "C" void kernel_launch(void* const* d_in, const int* in_sizes, int n_in,
                              void* d_out, int out_size, void* d_ws, size_t ws_size,
                              hipStream_t stream) {
    (void)in_sizes; (void)n_in; (void)out_size; (void)ws_size;
    const float* x_opt  = (const float*)d_in[0];
    const float* x_sar  = (const float*)d_in[1];
    const float* wq_opt = (const float*)d_in[2];
    const float* wk_opt = (const float*)d_in[3];
    const float* wv_opt = (const float*)d_in[4];
    const float* wq_sar = (const float*)d_in[5];
    const float* wk_sar = (const float*)d_in[6];
    const float* wv_sar = (const float*)d_in[7];
    const float* wp_opt = (const float*)d_in[8];
    const float* wp_sar = (const float*)d_in[9];
    const float* g_opt  = (const float*)d_in[10];
    const float* g_sar  = (const float*)d_in[11];
    float* out = (float*)d_out;

    const size_t QK = (size_t)NB * LSEQ * 32;   // 256K halves each
    const size_t VT = (size_t)NB * 64 * LSEQ;   // 512K halves each
    _Float16* ws  = (_Float16*)d_ws;            // total ~5 MB f16 scratch
    _Float16* Q1  = ws;
    _Float16* K1  = Q1 + QK;
    _Float16* Q2  = K1 + QK;
    _Float16* K2  = Q2 + QK;
    _Float16* Vt1 = K2 + QK;
    _Float16* Vt2 = Vt1 + VT;
    _Float16* St  = Vt2 + VT;

    qkv_kernel<<<dim3(LSEQ / 128, NB), 128, 0, stream>>>(
        x_opt, x_sar, wq_opt, wk_opt, wv_opt, wq_sar, wk_sar, wv_sar,
        Q1, K1, Vt1, Q2, K2);

    // pass 1: T^T = (softmax(q_sar k_sar) @ [v_opt|v_sar])^T
    attn_kernel<<<dim3(LSEQ / 64, NB), 128, 0, stream>>>(Q1, K1, Vt1, Vt2);
    // pass 2: S^T = (softmax(q_opt k_opt) @ T)^T
    attn_kernel<<<dim3(LSEQ / 64, NB), 128, 0, stream>>>(Q2, K2, Vt2, St);

    proj_kernel<<<dim3(LSEQ / 256, 64, NB), 256, 0, stream>>>(
        x_opt, x_sar, wp_opt, wp_sar, g_opt, g_sar, St, out);
}
